// CAttention_64974265254342
// MI455X (gfx1250) — matrix operand order repaired and checked by reading, compiled, not run on hardware
//
#include <hip/hip_runtime.h>
#include <hip/hip_bf16.h>

// Problem constants (from reference)
#define B_    4
#define NSEQ  2048
#define DIM_  1024
#define H_    8
#define HD_   128
#define SCALE_ 0.08838834764831845f   // 128^-0.5

typedef __attribute__((ext_vector_type(16))) __bf16 v16bf;
typedef __attribute__((ext_vector_type(4)))  __bf16 v4bf;
typedef __attribute__((ext_vector_type(8)))  float  v8f;
typedef __attribute__((ext_vector_type(4)))  unsigned int u32x4;
typedef __attribute__((ext_vector_type(8)))  int i32x8;
typedef __attribute__((ext_vector_type(4)))  int i32x4;

union FragU { v16bf v; uint4 u[2]; };

__device__ inline v8f wmma_bf16(v16bf a, v16bf b, v8f c) {
  // D = A(16x32 bf16) * B(32x16 bf16) + C(16x16 f32)
  return __builtin_amdgcn_wmma_f32_16x16x32_bf16(
      /*neg_a=*/false, a, /*neg_b=*/false, b,
      /*c_mod=*/(short)0, c, /*reuse_a=*/false, /*reuse_b=*/false);
}

// A-fragment (16x32, 16-bit): lane m = lane%16, 16 contiguous K values
// [k0,k0+16) where k0 = kbase + (lane/16)*16.  Two b128 loads.
__device__ inline v16bf load_afrag(const __bf16* __restrict__ row, int k0) {
  FragU f;
  f.u[0] = *reinterpret_cast<const uint4*>(row + k0);
  f.u[1] = *reinterpret_cast<const uint4*>(row + k0 + 8);
  return f.v;
}

// B-fragment (32x16, 16-bit): lane n = lane%16, 16 contiguous K values
// starting at kbase + (lane/16)*16.  Two b128 loads (global or LDS).
__device__ inline v16bf load_bfrag(const __bf16* p) {
  FragU f;
  f.u[0] = reinterpret_cast<const uint4*>(p)[0];
  f.u[1] = reinterpret_cast<const uint4*>(p)[1];
  return f.v;
}

// ---------------------------------------------------------------------------
// Tensor Data Mover: 2D tile load Global -> LDS (D# per CDNA5 ISA).
// dword0cfg carries data_size / pad_enable / pad_interval / pad_amount.
// This toolchain's builtin is the 6-arg form:
//   (u32x4 g0, i32x8 g1, i32x4 g2, i32x4 g3, i32x8 extra, i32 cpol)
// ---------------------------------------------------------------------------
__device__ inline void tdm_load_2d(const void* gaddr, unsigned lds_off,
                                   unsigned dword0cfg,
                                   unsigned tensor_d0, unsigned tensor_d1,
                                   unsigned tile_d0, unsigned tile_d1,
                                   unsigned long long stride0) {
  const unsigned long long ga = (unsigned long long)gaddr;
  u32x4 g0;
  g0[0] = 1u;                                             // count=1 (valid user D#)
  g0[1] = lds_off;                                        // lds_addr
  g0[2] = (unsigned)(ga & 0xFFFFFFFFu);                   // global_addr[31:0]
  g0[3] = (unsigned)((ga >> 32) & 0x01FFFFFFu) | (2u << 30); // addr[56:32] | type=2
  i32x8 g1;
  g1[0] = (int)dword0cfg;                                 // mask=0|data_size|pad cfg
  g1[1] = (int)((tensor_d0 & 0xFFFFu) << 16);             // [79:48] tensor_dim0 lo
  g1[2] = (int)((tensor_d0 >> 16) | ((tensor_d1 & 0xFFFFu) << 16));
  g1[3] = (int)((tensor_d1 >> 16) | (tile_d0 << 16));     // [127:112] tile_dim0
  g1[4] = (int)(tile_d1 & 0xFFFFu);                       // [143:128] tile_dim1
  g1[5] = (int)(unsigned)(stride0 & 0xFFFFFFFFu);         // tensor_dim0_stride lo
  g1[6] = (int)(unsigned)((stride0 >> 32) & 0xFFFFu);     // stride hi16
  g1[7] = 0;
  i32x4 z4; z4[0] = 0; z4[1] = 0; z4[2] = 0; z4[3] = 0;   // 2D: groups 2/3 unused
  i32x8 z8;
  #pragma unroll
  for (int i = 0; i < 8; ++i) z8[i] = 0;
  __builtin_amdgcn_tensor_load_to_lds(g0, g1, z4, z4, z8, 0);
}

__device__ inline unsigned lds_off_of(const void* p) {
  return (unsigned)(unsigned long long)p;                 // LDS offset = low 32 bits
}

// data_size=2B (code 1), pad_enable, pad_interval, pad_amount (codes per D#)
#define K_CFG ((1u << 16) | (1u << 20) | (5u << 22) | (3u << 25))  // 64dw rows +4dw pad
#define V_CFG ((1u << 16) | (1u << 20) | (3u << 22) | (3u << 25))  // 16dw rows +4dw pad

// ---------------------------------------------------------------------------
// One-time fp32 -> bf16 staging (vectorized: float4 in, 4x bf16 out).
// ---------------------------------------------------------------------------
__global__ __launch_bounds__(256)
void cvt_bf16_kernel(const float* __restrict__ in, __bf16* __restrict__ out, int n4) {
  const int i = blockIdx.x * 256 + threadIdx.x;
  if (i < n4) {
    const float4 f = reinterpret_cast<const float4*>(in)[i];
    v4bf o;
    o[0] = (__bf16)f.x; o[1] = (__bf16)f.y; o[2] = (__bf16)f.z; o[3] = (__bf16)f.w;
    reinterpret_cast<v4bf*>(out)[i] = o;
  }
}

// ---------------------------------------------------------------------------
// GEMM: C[M=8192, N=1024] = A[M,1024] @ W[1024,1024]^T, all-bf16 operands.
// Wave = 64x64 (4x4 WMMA tiles): 16 wmma per 32-k step vs 16 b128 loads.
// Block = 256 threads = 8 waves as 2 (rows) x 4 (cols) -> 128 x 256 per block.
// MODE 0: store bf16 head-major   out[((b*H+h)*NSEQ+tok)*HD + hd]      (Q,K)
// MODE 1: store bf16 transposed   out[((b*H+h)*HD+hd)*NSEQ + tok]      (V)
// MODE 2: store fp32 + bias       outf[row*DIM + col]                  (proj)
// ---------------------------------------------------------------------------
template <int MODE>
__global__ __launch_bounds__(256)
void wmma_gemm_kernel(const __bf16* __restrict__ A, const __bf16* __restrict__ W,
                      const float* __restrict__ bias,
                      __bf16* __restrict__ outb, float* __restrict__ outf) {
  const int lane = threadIdx.x & 31;
  const int wave = threadIdx.x >> 5;
  const int half = lane >> 4;
  const int l16  = lane & 15;
  const int m0 = blockIdx.y * 128 + (wave & 1) * 64;
  const int n0 = blockIdx.x * 256 + (wave >> 1) * 64;

  v8f acc[4][4] = {};

  for (int k = 0; k < DIM_; k += 32) {
    v16bf af[4], bfr[4];
    #pragma unroll
    for (int i = 0; i < 4; ++i)
      af[i] = load_afrag(A + (size_t)(m0 + i * 16 + l16) * DIM_, k + half * 16);
    #pragma unroll
    for (int j = 0; j < 4; ++j)
      bfr[j] = load_bfrag(W + (size_t)(n0 + j * 16 + l16) * DIM_ + k + half * 16);
    #pragma unroll
    for (int i = 0; i < 4; ++i)
      #pragma unroll
      for (int j = 0; j < 4; ++j)
        acc[i][j] = wmma_bf16(af[i], bfr[j], acc[i][j]);
  }

  #pragma unroll
  for (int i = 0; i < 4; ++i) {
    #pragma unroll
    for (int j = 0; j < 4; ++j) {
      const int gcol = n0 + j * 16 + l16;
      #pragma unroll
      for (int r = 0; r < 8; ++r) {
        const int grow = m0 + i * 16 + r + half * 8;   // C-layout: M = r + 8*half
        const float val = acc[i][j][r];
        if (MODE == 2) {
          outf[(size_t)grow * DIM_ + gcol] = val + bias[gcol];
        } else {
          const int b   = grow >> 11;        // / NSEQ
          const int tok = grow & (NSEQ - 1);
          const int h   = gcol >> 7;         // / HD
          const int hd  = gcol & (HD_ - 1);
          size_t idx;
          if (MODE == 0) idx = (((size_t)(b * H_ + h)) * NSEQ + tok) * HD_ + hd;
          else           idx = (((size_t)(b * H_ + h)) * HD_  + hd) * NSEQ + tok;
          outb[idx] = (__bf16)val;
        }
      }
    }
  }
}

// ---------------------------------------------------------------------------
// Flash-style gated attention. Block = 8 waves sharing one (b,h); each wave
// owns one 16-query tile. K/V 32-key slabs are staged into LDS by the Tensor
// Data Mover (double-buffered, TENSORcnt-pipelined), then consumed as WMMA
// fragments via bank-conflict-free padded ds_load_b128.
// Qh,Kh: [B,H,N,HD] bf16 ; Vt: [B,H,HD,N] bf16 ; trans: [B,N,N] f32
// X out: [B,N,DIM] bf16 (token-major for the final projection GEMM).
// ---------------------------------------------------------------------------
__global__ __launch_bounds__(256)
void attn_kernel(const __bf16* __restrict__ Qh, const __bf16* __restrict__ Kh,
                 const __bf16* __restrict__ Vt, const float* __restrict__ trans,
                 __bf16* __restrict__ X) {
  __shared__ alignas(16) __bf16 kbuf[2][32][136];   // 32 keys x 128 hd (+8 pad)
  __shared__ alignas(16) __bf16 vbuf[2][128][40];   // 128 hd x 32 keys (+8 pad)
  __shared__ alignas(16) __bf16 plds[8][16][32];    // per-wave P transpose tile

  const int lane = threadIdx.x & 31;
  const int wave = threadIdx.x >> 5;
  const int half = lane >> 4;
  const int l16  = lane & 15;
  const int h = blockIdx.y;
  const int b = blockIdx.z;
  const int q0 = (blockIdx.x * 8 + wave) * 16;

  const size_t bh = (size_t)(b * H_ + h);
  const __bf16* Qp = Qh + bh * (size_t)NSEQ * HD_;
  const __bf16* Kp = Kh + bh * (size_t)NSEQ * HD_;
  const __bf16* Vp = Vt + bh * (size_t)HD_ * NSEQ;
  const float*  Tp = trans + (size_t)b * NSEQ * NSEQ;

  // Q A-fragments over hd (4 chunks of 32), resident for the whole key sweep.
  v16bf qf[4];
  {
    const __bf16* qrow = Qp + (size_t)(q0 + l16) * HD_;
    #pragma unroll
    for (int c = 0; c < 4; ++c) qf[c] = load_afrag(qrow, c * 32 + half * 16);
  }

  v8f o[8] = {};                        // 16 x 128 output accumulator (C-layout)
  float mrow[8], drow[8];
  #pragma unroll
  for (int r = 0; r < 8; ++r) { mrow[r] = -1e30f; drow[r] = 0.f; }

  const int nIter = NSEQ / 32;          // 64 key blocks

  // Prologue: TDM-stage slab 0 (K: 32x128 tile; V: 128x32 tile).
  if (wave == 0) {
    tdm_load_2d(Kp, lds_off_of(&kbuf[0][0][0]), K_CFG,
                HD_, NSEQ, /*tile*/ HD_, 32, /*stride0*/ HD_);
    tdm_load_2d(Vp, lds_off_of(&vbuf[0][0][0]), V_CFG,
                NSEQ, HD_, /*tile*/ 32, HD_, /*stride0*/ NSEQ);
  }

  for (int it = 0; it < nIter; ++it) {
    const int kb  = it * 32;
    const int cur = it & 1;

    if (wave == 0) {
      if (it + 1 < nIter) {               // stage next slab into the other buffer
        const int nxt = cur ^ 1;
        tdm_load_2d(Kp + (size_t)(kb + 32) * HD_, lds_off_of(&kbuf[nxt][0][0]),
                    K_CFG, HD_, NSEQ, HD_, 32, HD_);
        tdm_load_2d(Vp + (kb + 32), lds_off_of(&vbuf[nxt][0][0]),
                    V_CFG, NSEQ, HD_, 32, HD_, NSEQ);
        __builtin_amdgcn_s_wait_tensorcnt(2);   // current slab landed
      } else {
        __builtin_amdgcn_s_wait_tensorcnt(0);
      }
    }
    __syncthreads();                       // slab `cur` visible to all waves

    // ---- scores: two 16x16 tiles over keys [kb,kb+16) and [kb+16,kb+32)
    v8f s0 = {}, s1 = {};
    #pragma unroll
    for (int c = 0; c < 4; ++c) {
      v16bf bk0 = load_bfrag(&kbuf[cur][l16][c * 32 + half * 16]);
      v16bf bk1 = load_bfrag(&kbuf[cur][16 + l16][c * 32 + half * 16]);
      s0 = wmma_bf16(qf[c], bk0, s0);
      s1 = wmma_bf16(qf[c], bk1, s1);
    }

    // ---- gate by trans_mat, online softmax update (denominator over ALL keys,
    //      numerator excludes gated==0, matching the reference semantics).
    float alpha[8];
    #pragma unroll
    for (int r = 0; r < 8; ++r) {
      const int qrow = q0 + r + half * 8;
      const float t0 = Tp[(size_t)qrow * NSEQ + kb + l16];
      const float t1 = Tp[(size_t)qrow * NSEQ + kb + 16 + l16];
      const float g0 = s0[r] * SCALE_ * t0;
      const float g1 = s1[r] * SCALE_ * t1;
      float mx = fmaxf(g0, g1);
      #pragma unroll
      for (int m = 1; m < 16; m <<= 1) mx = fmaxf(mx, __shfl_xor(mx, m, 16));
      const float mnew = fmaxf(mrow[r], mx);
      alpha[r] = __expf(mrow[r] - mnew);
      mrow[r] = mnew;
      const float e0 = __expf(g0 - mnew);
      const float e1 = __expf(g1 - mnew);
      float es = e0 + e1;
      #pragma unroll
      for (int m = 1; m < 16; m <<= 1) es += __shfl_xor(es, m, 16);
      drow[r] = drow[r] * alpha[r] + es;
      plds[wave][r + half * 8][l16]      = (__bf16)((g0 != 0.f) ? e0 : 0.f);
      plds[wave][r + half * 8][16 + l16] = (__bf16)((g1 != 0.f) ? e1 : 0.f);
    }
    __builtin_amdgcn_wave_barrier();
    asm volatile("s_wait_dscnt 0" ::: "memory");   // per-wave LDS RAW fence

    // ---- rescale running output by alpha
    #pragma unroll
    for (int t = 0; t < 8; ++t)
      #pragma unroll
      for (int r = 0; r < 8; ++r) o[t][r] *= alpha[r];

    // ---- P tile back as A-fragment (16x32 over keys), then P @ V from LDS
    FragU pf;
    pf.u[0] = *reinterpret_cast<const uint4*>(&plds[wave][l16][half * 16]);
    pf.u[1] = *reinterpret_cast<const uint4*>(&plds[wave][l16][half * 16 + 8]);
    #pragma unroll
    for (int t = 0; t < 8; ++t) {
      v16bf bv = load_bfrag(&vbuf[cur][t * 16 + l16][half * 16]);
      o[t] = wmma_bf16(pf.v, bv, o[t]);
    }
    __syncthreads();                       // all reads of slab `cur` done before
  }                                        // wave0 overwrites it next iteration

  // ---- normalize and store token-major bf16
  #pragma unroll
  for (int t = 0; t < 8; ++t) {
    #pragma unroll
    for (int r = 0; r < 8; ++r) {
      const int qrow = q0 + r + half * 8;
      const float val = o[t][r] / drow[r];
      X[((size_t)(b * NSEQ + qrow)) * DIM_ + h * HD_ + t * 16 + l16] = (__bf16)val;
    }
  }
}

// ---------------------------------------------------------------------------
extern "C" void kernel_launch(void* const* d_in, const int* in_sizes, int n_in,
                              void* d_out, int out_size, void* d_ws, size_t ws_size,
                              hipStream_t stream) {
  (void)in_sizes; (void)n_in; (void)out_size; (void)ws_size;
  const float* q     = (const float*)d_in[0];
  const float* k     = (const float*)d_in[1];
  const float* v     = (const float*)d_in[2];
  const float* trans = (const float*)d_in[3];
  const float* Wq    = (const float*)d_in[4];
  const float* Wk    = (const float*)d_in[5];
  const float* Wv    = (const float*)d_in[6];
  const float* Wp    = (const float*)d_in[7];
  const float* bp    = (const float*)d_in[8];
  float* out = (float*)d_out;

  const size_t TOK = (size_t)B_ * NSEQ * DIM_;       // 8M elements
  const size_t WEL = (size_t)DIM_ * DIM_;            // 1M elements
  __bf16* Qh  = (__bf16*)d_ws;                       // head-major Q
  __bf16* Kh  = Qh + TOK;                            // head-major K
  __bf16* Vt  = Kh + TOK;                            // transposed V
  __bf16* X   = Vt + TOK;                            // attention output
  __bf16* qb  = X  + TOK;                            // bf16 staged inputs
  __bf16* kb  = qb + TOK;
  __bf16* vb  = kb + TOK;
  __bf16* Wqb = vb + TOK;                            // bf16 staged weights
  __bf16* Wkb = Wqb + WEL;
  __bf16* Wvb = Wkb + WEL;
  __bf16* Wpb = Wvb + WEL;                           // total: 120 MB

  // ---- stage everything to bf16 once (removes cvt VALU from GEMM hot loops)
  {
    const int tok4 = (int)(TOK / 4), w4 = (int)(WEL / 4);
    const dim3 cb(256);
    cvt_bf16_kernel<<<dim3((tok4 + 255) / 256), cb, 0, stream>>>(q,  qb,  tok4);
    cvt_bf16_kernel<<<dim3((tok4 + 255) / 256), cb, 0, stream>>>(k,  kb,  tok4);
    cvt_bf16_kernel<<<dim3((tok4 + 255) / 256), cb, 0, stream>>>(v,  vb,  tok4);
    cvt_bf16_kernel<<<dim3((w4   + 255) / 256), cb, 0, stream>>>(Wq, Wqb, w4);
    cvt_bf16_kernel<<<dim3((w4   + 255) / 256), cb, 0, stream>>>(Wk, Wkb, w4);
    cvt_bf16_kernel<<<dim3((w4   + 255) / 256), cb, 0, stream>>>(Wv, Wvb, w4);
    cvt_bf16_kernel<<<dim3((w4   + 255) / 256), cb, 0, stream>>>(Wp, Wpb, w4);
  }

  const dim3 gemm_grid(DIM_ / 256, (B_ * NSEQ) / 128);  // 4 x 64
  const dim3 gemm_block(256);

  // Q/K/V projections (bf16 in, bf16 head-layouts out)
  wmma_gemm_kernel<0><<<gemm_grid, gemm_block, 0, stream>>>(qb, Wqb, nullptr, Qh, nullptr);
  wmma_gemm_kernel<0><<<gemm_grid, gemm_block, 0, stream>>>(kb, Wkb, nullptr, Kh, nullptr);
  wmma_gemm_kernel<1><<<gemm_grid, gemm_block, 0, stream>>>(vb, Wvb, nullptr, Vt, nullptr);

  // Gated-softmax attention (TDM-staged K/V slabs)
  const dim3 attn_grid(NSEQ / 16 / 8, H_, B_);          // 16 x 8 x 4
  attn_kernel<<<attn_grid, dim3(256), 0, stream>>>(Qh, Kh, Vt, trans, X);

  // Output projection + bias (bf16 in, fp32 out)
  wmma_gemm_kernel<2><<<gemm_grid, gemm_block, 0, stream>>>(X, Wpb, bp, nullptr, out);
}
